// SlidingWindowAttention_66915590472349
// MI455X (gfx1250) — compile-verified, matrix operand order checked by reading, not run on hardware
//
#include <hip/hip_runtime.h>
#include <hip/hip_bf16.h>
#include <math.h>

typedef __attribute__((ext_vector_type(16))) __bf16 v16bf;
typedef __attribute__((ext_vector_type(8)))  float  v8f;

static constexpr int BB = 2;
static constexpr int SS = 2048;
static constexpr int HH = 2048;
static constexpr int NH = 16;
static constexpr int HD = 128;
static constexpr int MM = BB * SS;   // 4096 rows

// ---------------------------------------------------------------------------
// CDNA5 async global->LDS copy (ASYNCcnt) helpers
// ---------------------------------------------------------------------------
__device__ __forceinline__ void async_b128(__bf16* lds_dst, const __bf16* gsrc) {
    unsigned lds_off = (unsigned)(size_t)lds_dst;   // low 32 bits of generic ptr = LDS addr
    asm volatile("global_load_async_to_lds_b128 %0, %1, off"
                 :: "v"(lds_off), "v"(gsrc) : "memory");
}
__device__ __forceinline__ void wait_async0() {
    asm volatile("s_wait_asynccnt 0" ::: "memory");
}

// ---------------------------------------------------------------------------
// fp32 -> bf16 conversion
// ---------------------------------------------------------------------------
__global__ __launch_bounds__(256) void f32_to_bf16_kernel(const float* __restrict__ in,
                                                          __bf16* __restrict__ out,
                                                          size_t n) {
    size_t i = (size_t)blockIdx.x * blockDim.x + threadIdx.x;
    if (i < n) out[i] = (__bf16)in[i];
}

// ---------------------------------------------------------------------------
// C[M,N] = A[M,K] * Bm[N,K]^T  (torch Linear), bf16 in, f32 accum.
// Block tile 128x256, BK=32, 256 threads (8 waves = 2M x 4N), wave tile 64x64.
// Double-buffered LDS (offset-selected, so fragment reads stay ds_load) fed by
// async-to-LDS copies. VTRANS: scatter output as V^T into [B, NH, HD, S].
// ---------------------------------------------------------------------------
__device__ __forceinline__ void store_out(__bf16* p, float v) { *p = (__bf16)v; }
__device__ __forceinline__ void store_out(float*  p, float v) { *p = v; }

template <typename OutT, bool VTRANS>
__global__ __launch_bounds__(256) void gemm_abt_kernel(const __bf16* __restrict__ A,
                                                       const __bf16* __restrict__ Bm,
                                                       OutT* __restrict__ C,
                                                       int M, int N, int K) {
    constexpr int LDT   = 40;                     // padded LDS stride (elems)
    constexpr int ASZ   = 128 * LDT;              // one A buffer (elems)
    constexpr int BSZ   = 256 * LDT;              // one B buffer (elems)
    __shared__ __bf16 As[2 * ASZ];                // 2 x 10240 B
    __shared__ __bf16 Bs[2 * BSZ];                // 2 x 20480 B

    const int tid   = threadIdx.x;
    const int wave  = tid >> 5;
    const int lane  = tid & 31;
    const int fr    = lane & 15;
    const int kq    = (lane >> 4) << 4;           // 0/16 (K sub-offset)
    const int rhalf = (lane >> 4) << 3;           // 0/8  (C row sub-offset)
    const int waveM = wave >> 2;                  // 0..1 -> 64 rows
    const int waveN = wave & 3;                   // 0..3 -> 64 cols
    const int m0    = blockIdx.y * 128;
    const int n0    = blockIdx.x * 256;

    const int arow = tid >> 1, ah = tid & 1;      // A loader: 16 elems/thread

    auto issue_tile = [&](int buf, int k0) {
        const __bf16* ga = A + (size_t)(m0 + arow) * K + k0 + ah * 16;
        __bf16* da = &As[buf * ASZ + arow * LDT + ah * 16];
        async_b128(da,     ga);
        async_b128(da + 8, ga + 8);
        const __bf16* gb = Bm + (size_t)(n0 + tid) * K + k0;   // 32 elems/thread
        __bf16* db = &Bs[buf * BSZ + tid * LDT];
#pragma unroll
        for (int i = 0; i < 4; ++i) async_b128(db + i * 8, gb + i * 8);
    };

    v8f acc[4][4];
    const v8f vz = {0.f, 0.f, 0.f, 0.f, 0.f, 0.f, 0.f, 0.f};
#pragma unroll
    for (int mi = 0; mi < 4; ++mi)
#pragma unroll
        for (int ni = 0; ni < 4; ++ni) acc[mi][ni] = vz;

    issue_tile(0, 0);
    wait_async0();
    __syncthreads();

    for (int k0 = 0; k0 < K; k0 += 32) {
        const int cur  = (k0 >> 5) & 1;
        const int aoff = cur * ASZ;
        const int boff = cur * BSZ;
        if (k0 + 32 < K) issue_tile(cur ^ 1, k0 + 32);   // overlap with compute

        v16bf af[4], bf[4];
#pragma unroll
        for (int mi = 0; mi < 4; ++mi)
            af[mi] = *(const v16bf*)&As[aoff + (waveM * 64 + mi * 16 + fr) * LDT + kq];
#pragma unroll
        for (int ni = 0; ni < 4; ++ni)
            bf[ni] = *(const v16bf*)&Bs[boff + (waveN * 64 + ni * 16 + fr) * LDT + kq];
#pragma unroll
        for (int mi = 0; mi < 4; ++mi)
#pragma unroll
            for (int ni = 0; ni < 4; ++ni)
                acc[mi][ni] = __builtin_amdgcn_wmma_f32_16x16x32_bf16(
                    false, af[mi], false, bf[ni], (short)0, acc[mi][ni], false, false);

        wait_async0();
        __syncthreads();
    }

#pragma unroll
    for (int mi = 0; mi < 4; ++mi)
#pragma unroll
        for (int ni = 0; ni < 4; ++ni)
#pragma unroll
            for (int r = 0; r < 8; ++r) {
                int row = m0 + waveM * 64 + mi * 16 + r + rhalf;
                int col = n0 + waveN * 64 + ni * 16 + fr;
                if constexpr (VTRANS) {
                    int b = row >> 11, s = row & (SS - 1);
                    int h = col >> 7,  d = col & (HD - 1);
                    C[(((size_t)b * NH + h) * HD + d) * SS + s] = (OutT)acc[mi][ni][r];
                } else {
                    store_out(&C[(size_t)row * N + col], acc[mi][ni][r]);
                }
            }
}

// ---------------------------------------------------------------------------
// RoPE (HF rotate_half) in place on Q and K (bf16)
// ---------------------------------------------------------------------------
__global__ __launch_bounds__(256) void rope_kernel(__bf16* __restrict__ Q,
                                                   __bf16* __restrict__ K,
                                                   const float* __restrict__ cosT,
                                                   const float* __restrict__ sinT,
                                                   const long long* __restrict__ pos) {
    size_t idx = (size_t)blockIdx.x * blockDim.x + threadIdx.x;  // B*S*NH*64
    if (idx >= (size_t)MM * NH * 64) return;
    int i      = idx & 63;
    int h      = (int)((idx >> 6) & (NH - 1));
    size_t tok = idx >> 10;                     // b*S + s
    int b      = (int)(tok >> 11);
    int s      = (int)(tok & (SS - 1));
    long long p = pos[(size_t)b * SS + s];
    const float c1 = cosT[p * HD + i],      sn1 = sinT[p * HD + i];
    const float c2 = cosT[p * HD + i + 64], sn2 = sinT[p * HD + i + 64];
    size_t base = tok * HH + (size_t)h * HD;
    float q1 = (float)Q[base + i], q2 = (float)Q[base + i + 64];
    Q[base + i]      = (__bf16)(q1 * c1 - q2 * sn1);
    Q[base + i + 64] = (__bf16)(q2 * c2 + q1 * sn2);
    float k1 = (float)K[base + i], k2 = (float)K[base + i + 64];
    K[base + i]      = (__bf16)(k1 * c1 - k2 * sn1);
    K[base + i + 64] = (__bf16)(k2 * c2 + k1 * sn2);
}

// ---------------------------------------------------------------------------
// Flash attention. Q/K: [B*S, H]; V already transposed: Vt[b][h][d][s].
// 128 q-rows per block, 64-col KV chunks, double-buffered async LDS staging
// selected purely by offsets so fragment reads stay on the ds_load path.
// ---------------------------------------------------------------------------
__global__ __launch_bounds__(256) void flash_attn_kernel(const __bf16* __restrict__ Q,
                                                         const __bf16* __restrict__ Kb,
                                                         const __bf16* __restrict__ Vt_g,
                                                         const float* __restrict__ amask,
                                                         __bf16* __restrict__ Out) {
    constexpr int QLD = 136, KLD = 136, VLD = 72, PLD = 72;
    constexpr int KSZ = 64 * KLD;                 // one K buffer (elems)
    constexpr int VSZ = 128 * VLD;                // one V buffer (elems)
    extern __shared__ char smem[];
    __bf16* Qs = (__bf16*)smem;                   // 128 x QLD
    __bf16* Ks = Qs + 128 * QLD;                  // 2 x KSZ
    __bf16* Vt = Ks + 2 * KSZ;                    // 2 x VSZ (dim-major)
    __bf16* Ps = Vt + 2 * VSZ;                    // 128 x PLD

    const int tid   = threadIdx.x;
    const int wave  = tid >> 5;
    const int lane  = tid & 31;
    const int fr    = lane & 15;
    const int kq    = (lane >> 4) << 4;
    const int rhalf = (lane >> 4) << 3;
    const int q0    = blockIdx.x * 128;
    const int h     = blockIdx.y;
    const int b     = blockIdx.z;
    const int rbase = wave * 16;
    const float scale = 0.08838834764831845f;     // 1/sqrt(128)

    const __bf16* vbase = Vt_g + ((size_t)b * NH + h) * HD * SS;

    auto issue_chunk = [&](int buf, int kc) {
        // K: 64 rows x 128 elems; 4 async b128 per thread
        const int krow = tid >> 2, kqt = tid & 3;
        const __bf16* gk = Kb + (size_t)(b * SS + kc + krow) * HH + h * HD + kqt * 32;
        __bf16* dk = &Ks[buf * KSZ + krow * KLD + kqt * 32];
#pragma unroll
        for (int i = 0; i < 4; ++i) async_b128(dk + i * 8, gk + i * 8);
        // V^T: 128 dim-rows x 64 seq elems; 4 async b128 per thread
        const int vrow = tid >> 1, vh = tid & 1;
        const __bf16* gv = vbase + (size_t)vrow * SS + kc + vh * 32;
        __bf16* dv = &Vt[buf * VSZ + vrow * VLD + vh * 32];
#pragma unroll
        for (int i = 0; i < 4; ++i) async_b128(dv + i * 8, gv + i * 8);
    };

    // prologue: resident Q tile + chunk 0, all via async copies
    {
        const int lrow = tid >> 1, lh = tid & 1;   // 64 elems/thread
        const __bf16* gq = Q + (size_t)(b * SS + q0 + lrow) * HH + h * HD + lh * 64;
        __bf16* dq = &Qs[lrow * QLD + lh * 64];
#pragma unroll
        for (int i = 0; i < 8; ++i) async_b128(dq + i * 8, gq + i * 8);
    }
    issue_chunk(0, 0);
    wait_async0();
    __syncthreads();

    // Q fragments are invariant across KV chunks: load once, keep in VGPRs.
    v16bf qf[4];
#pragma unroll
    for (int kk = 0; kk < 4; ++kk)
        qf[kk] = *(const v16bf*)&Qs[(rbase + fr) * QLD + kk * 32 + kq];

    float m_i[8], l_i[8];
    v8f oacc[8];
    const v8f vz = {0.f, 0.f, 0.f, 0.f, 0.f, 0.f, 0.f, 0.f};
#pragma unroll
    for (int r = 0; r < 8; ++r) { m_i[r] = -INFINITY; l_i[r] = 0.f; }
#pragma unroll
    for (int i = 0; i < 8; ++i) oacc[i] = vz;

    for (int kc = 0; kc < SS; kc += 64) {
        const int cur  = (kc >> 6) & 1;
        const int koff = cur * KSZ;
        const int voff = cur * VSZ;
        if (kc + 64 < SS) issue_chunk(cur ^ 1, kc + 64);   // overlap with compute

        // ---- S = Q K^T : 16 rows/wave x 64 cols, K-dim 128 ----
        v8f sc[4];
#pragma unroll
        for (int nf = 0; nf < 4; ++nf) sc[nf] = vz;
#pragma unroll
        for (int kk = 0; kk < 4; ++kk) {
#pragma unroll
            for (int nf = 0; nf < 4; ++nf) {
                v16bf kf = *(const v16bf*)&Ks[koff + (nf * 16 + fr) * KLD + kk * 32 + kq];
                sc[nf] = __builtin_amdgcn_wmma_f32_16x16x32_bf16(
                    false, qf[kk], false, kf, (short)0, sc[nf], false, false);
            }
        }

        // ---- scale + mask + online softmax (rows live in 16-lane groups) ----
        float rmax[8];
#pragma unroll
        for (int r = 0; r < 8; ++r) rmax[r] = -INFINITY;
#pragma unroll
        for (int nf = 0; nf < 4; ++nf) {
            float add = (1.0f - amask[(size_t)b * SS + kc + nf * 16 + fr]) * -10000.0f;
#pragma unroll
            for (int r = 0; r < 8; ++r) {
                float v = sc[nf][r] * scale + add;
                sc[nf][r] = v;
                rmax[r] = fmaxf(rmax[r], v);
            }
        }
#pragma unroll
        for (int r = 0; r < 8; ++r)
#pragma unroll
            for (int off = 1; off < 16; off <<= 1)
                rmax[r] = fmaxf(rmax[r], __shfl_xor(rmax[r], off, 16));

        float rsum[8];
#pragma unroll
        for (int r = 0; r < 8; ++r) {
            float mnew  = fmaxf(m_i[r], rmax[r]);
            float alpha = __expf(m_i[r] - mnew);
            m_i[r] = mnew;
            l_i[r] *= alpha;
            rsum[r] = 0.f;
#pragma unroll
            for (int of = 0; of < 8; ++of) oacc[of][r] *= alpha;
        }
        // P = exp(S - m) -> LDS bf16 (wave-private rows; LDS in-order per wave)
#pragma unroll
        for (int nf = 0; nf < 4; ++nf)
#pragma unroll
            for (int r = 0; r < 8; ++r) {
                float p = __expf(sc[nf][r] - m_i[r]);
                rsum[r] += p;
                Ps[(rbase + r + rhalf) * PLD + nf * 16 + fr] = (__bf16)p;
            }
#pragma unroll
        for (int r = 0; r < 8; ++r) {
#pragma unroll
            for (int off = 1; off < 16; off <<= 1)
                rsum[r] += __shfl_xor(rsum[r], off, 16);
            l_i[r] += rsum[r];
        }

        // ---- O += P V : 16 rows x 128 dims, K-dim 64 ----
#pragma unroll
        for (int kk = 0; kk < 2; ++kk) {
            v16bf pf = *(const v16bf*)&Ps[(rbase + fr) * PLD + kk * 32 + kq];
#pragma unroll
            for (int of = 0; of < 8; ++of) {
                v16bf vf = *(const v16bf*)&Vt[voff + (of * 16 + fr) * VLD + kk * 32 + kq];
                oacc[of] = __builtin_amdgcn_wmma_f32_16x16x32_bf16(
                    false, pf, false, vf, (short)0, oacc[of], false, false);
            }
        }

        wait_async0();      // next chunk's async copies must land before swap
        __syncthreads();
    }

    // ---- normalize and write attn output (bf16, [B*S, H]) ----
    float rinv[8];
#pragma unroll
    for (int r = 0; r < 8; ++r) rinv[r] = 1.0f / l_i[r];
#pragma unroll
    for (int of = 0; of < 8; ++of)
#pragma unroll
        for (int r = 0; r < 8; ++r) {
            float o = oacc[of][r] * rinv[r];
            size_t row = (size_t)(b * SS + q0 + rbase + r + rhalf);
            Out[row * HH + h * HD + of * 16 + fr] = (__bf16)o;
        }
}

// ---------------------------------------------------------------------------
// launcher
// ---------------------------------------------------------------------------
extern "C" void kernel_launch(void* const* d_in, const int* in_sizes, int n_in,
                              void* d_out, int out_size, void* d_ws, size_t ws_size,
                              hipStream_t stream) {
    const float* hs   = (const float*)d_in[0];
    const float* wq   = (const float*)d_in[1];
    const float* wk   = (const float*)d_in[2];
    const float* wv   = (const float*)d_in[3];
    const float* wo   = (const float*)d_in[4];
    const float* cosT = (const float*)d_in[5];
    const float* sinT = (const float*)d_in[6];
    const float* amsk = (const float*)d_in[7];
    const long long* pos = (const long long*)d_in[8];
    float* out = (float*)d_out;

    // workspace carve (bf16): X, 4 weights, Q, K, V^T, attn  (~117 MB)
    __bf16* Xb  = (__bf16*)d_ws;
    __bf16* Wqb = Xb  + (size_t)MM * HH;
    __bf16* Wkb = Wqb + (size_t)HH * HH;
    __bf16* Wvb = Wkb + (size_t)HH * HH;
    __bf16* Wob = Wvb + (size_t)HH * HH;
    __bf16* Qb  = Wob + (size_t)HH * HH;
    __bf16* Kb  = Qb  + (size_t)MM * HH;
    __bf16* Vtb = Kb  + (size_t)MM * HH;   // [B, NH, HD, S]
    __bf16* Ab  = Vtb + (size_t)MM * HH;

    const size_t nX = (size_t)MM * HH;
    const size_t nW = (size_t)HH * HH;
    f32_to_bf16_kernel<<<(unsigned)((nX + 255) / 256), 256, 0, stream>>>(hs, Xb, nX);
    f32_to_bf16_kernel<<<(unsigned)((nW + 255) / 256), 256, 0, stream>>>(wq, Wqb, nW);
    f32_to_bf16_kernel<<<(unsigned)((nW + 255) / 256), 256, 0, stream>>>(wk, Wkb, nW);
    f32_to_bf16_kernel<<<(unsigned)((nW + 255) / 256), 256, 0, stream>>>(wv, Wvb, nW);
    f32_to_bf16_kernel<<<(unsigned)((nW + 255) / 256), 256, 0, stream>>>(wo, Wob, nW);

    dim3 gg(HH / 256, MM / 128);             // 8 x 32 = 256 blocks
    gemm_abt_kernel<__bf16, false><<<gg, 256, 0, stream>>>(Xb, Wqb, Qb, MM, HH, HH);
    gemm_abt_kernel<__bf16, false><<<gg, 256, 0, stream>>>(Xb, Wkb, Kb, MM, HH, HH);
    gemm_abt_kernel<__bf16, true ><<<gg, 256, 0, stream>>>(Xb, Wvb, Vtb, MM, HH, HH);

    const size_t nR = (size_t)MM * NH * 64;
    rope_kernel<<<(unsigned)((nR + 255) / 256), 256, 0, stream>>>(Qb, Kb, cosT, sinT, pos);

    dim3 fg(SS / 128, NH, BB);               // 16 x 16 x 2
    const size_t smem = (size_t)(128 * 136 + 2 * 64 * 136 + 2 * 128 * 72 + 128 * 72) * sizeof(__bf16);
    flash_attn_kernel<<<fg, 256, smem, stream>>>(Qb, Kb, Vtb, amsk, Ab);

    gemm_abt_kernel<float, false><<<gg, 256, 0, stream>>>(Ab, Wob, out, MM, HH, HH);
}
